// temporal_inter_attention_59768764891349
// MI455X (gfx1250) — matrix operand order repaired and checked by reading, not compiled, not run on hardware
//
#include <hip/hip_runtime.h>
#include <hip/hip_bf16.h>

// ---------------------------------------------------------------------------
// CDNA5 (gfx1250) fused temporal attention block, bf16 WMMA everywhere.
// Five GEMM kernels; all matmuls run through v_wmma_f32_16x16x32_bf16.
// Wave tile 64x64 (4x4 wmma tiles, 16 wmma : 16 ds_load_b128 per K-step),
// block tile 128x256 (8 wave32), v_perm_b32 transpose packing and packed
// f32->bf16 conversion in staging, global_prefetch of next tiles.
// ---------------------------------------------------------------------------

typedef __attribute__((ext_vector_type(8)))  float  v8f;
typedef __attribute__((ext_vector_type(2)))  float  v2f;
typedef __attribute__((ext_vector_type(2)))  __bf16 v2bf;
typedef __attribute__((ext_vector_type(8)))  __bf16 v8bf;
typedef __attribute__((ext_vector_type(16))) __bf16 v16bf;

#define LDP 40   // LDS row pitch in bf16 elements (80B: 16B aligned, bank-spread)

__device__ __forceinline__ unsigned short f2bf(float f) {
    union { __bf16 h; unsigned short u; } c;
    c.h = (__bf16)f;
    return c.u;
}
// pack two floats -> two bf16 in one dword (v_cvt_pk_bf16_f32 path)
__device__ __forceinline__ unsigned pack2(float a, float b) {
    v2f f = { a, b };
    union { v2bf h; unsigned u; } c;
    c.h = __builtin_convertvector(f, v2bf);
    return c.u;
}
// v_perm_b32 selectors: build {lo16(b),lo16(a)} / {hi16(b),hi16(a)}
__device__ __forceinline__ unsigned perm_lo(unsigned a_hi, unsigned b_lo) {
    return __builtin_amdgcn_perm(a_hi, b_lo, 0x05040100u);  // D = {a[15:0], b[15:0]}
}
__device__ __forceinline__ unsigned perm_hi(unsigned a_hi, unsigned b_lo) {
    return __builtin_amdgcn_perm(a_hi, b_lo, 0x07060302u);  // D = {a[31:16], b[31:16]}
}

// ---- staging: A tile 128 x 32 (rows = M, contiguous K), f32 src -> bf16 ----
__device__ __forceinline__ void stage_mk_f32(unsigned short* lds, const float* src,
                                             long ld, int row0, int Mlim, int tid) {
    const int r  = tid >> 1;          // 0..127
    const int kk = (tid & 1) << 4;    // 0 / 16
    const int row = row0 + r;
    uint4* dst = (uint4*)(lds + r * LDP + kk);
    if (row < Mlim) {
        const float4* p = (const float4*)(src + (long)row * ld + kk);
        float4 f0 = p[0], f1 = p[1], f2 = p[2], f3 = p[3];
        dst[0] = make_uint4(pack2(f0.x, f0.y), pack2(f0.z, f0.w),
                            pack2(f1.x, f1.y), pack2(f1.z, f1.w));
        dst[1] = make_uint4(pack2(f2.x, f2.y), pack2(f2.z, f2.w),
                            pack2(f3.x, f3.y), pack2(f3.z, f3.w));
    } else {
        dst[0] = make_uint4(0, 0, 0, 0);
        dst[1] = make_uint4(0, 0, 0, 0);
    }
}

// ---- staging: ROWS x 32 tile (contiguous K), bf16 source (straight copy) ----
template <int ROWS>
__device__ __forceinline__ void stage_mk_bf16(unsigned short* lds, const unsigned short* src,
                                              long ld, int row0, int Mlim, int tid) {
#pragma unroll
    for (int rr = 0; rr < ROWS / 128; ++rr) {
        const int r  = (tid >> 1) + rr * 128;
        const int kk = (tid & 1) << 4;
        const int row = row0 + r;
        uint4* dst = (uint4*)(lds + r * LDP + kk);
        if (row < Mlim) {
            const uint4* p = (const uint4*)(src + (long)row * ld + kk);
            dst[0] = p[0];
            dst[1] = p[1];
        } else {
            dst[0] = make_uint4(0, 0, 0, 0);
            dst[1] = make_uint4(0, 0, 0, 0);
        }
    }
}

// ---- staging: B tile from [k][n] source (K=32 rows, N=256 contiguous cols),
//      transposed into LDS [n][k]. Thread owns a 4(k) x 8(n) micro-block:
//      8 float4 loads -> packed cvt -> 8 x ds_store_b64. ----
__device__ __forceinline__ void stage_kn_f32(unsigned short* lds, const float* src,
                                             long ld, int tid) {
    const int kr = (tid >> 5) << 2;   // 0,4,..,28
    const int n0 = (tid & 31) << 3;   // 0,8,..,248
    float4 a[4][2];
#pragma unroll
    for (int j = 0; j < 4; ++j) {
        const float4* p = (const float4*)(src + (long)(kr + j) * ld + n0);
        a[j][0] = p[0];
        a[j][1] = p[1];
    }
#pragma unroll
    for (int half = 0; half < 2; ++half) {
        float4 r0 = a[0][half], r1 = a[1][half], r2 = a[2][half], r3 = a[3][half];
        const int nb = n0 + half * 4;
        *(uint2*)(lds + (nb + 0) * LDP + kr) = make_uint2(pack2(r0.x, r1.x), pack2(r2.x, r3.x));
        *(uint2*)(lds + (nb + 1) * LDP + kr) = make_uint2(pack2(r0.y, r1.y), pack2(r2.y, r3.y));
        *(uint2*)(lds + (nb + 2) * LDP + kr) = make_uint2(pack2(r0.z, r1.z), pack2(r2.z, r3.z));
        *(uint2*)(lds + (nb + 3) * LDP + kr) = make_uint2(pack2(r0.w, r1.w), pack2(r2.w, r3.w));
    }
}

// ---- same, bf16 source: v_perm_b32 register transpose, 8 x ds_store_b64 ----
__device__ __forceinline__ void stage_kn_bf16(unsigned short* lds, const unsigned short* src,
                                              long ld, int tid) {
    const int kr = (tid >> 5) << 2;
    const int n0 = (tid & 31) << 3;
    uint4 r0 = *(const uint4*)(src + (long)(kr + 0) * ld + n0);
    uint4 r1 = *(const uint4*)(src + (long)(kr + 1) * ld + n0);
    uint4 r2 = *(const uint4*)(src + (long)(kr + 2) * ld + n0);
    uint4 r3 = *(const uint4*)(src + (long)(kr + 3) * ld + n0);
    unsigned w0[4] = { r0.x, r0.y, r0.z, r0.w };
    unsigned w1[4] = { r1.x, r1.y, r1.z, r1.w };
    unsigned w2[4] = { r2.x, r2.y, r2.z, r2.w };
    unsigned w3[4] = { r3.x, r3.y, r3.z, r3.w };
#pragma unroll
    for (int j = 0; j < 4; ++j) {   // word j holds cols n0+2j, n0+2j+1
        *(uint2*)(lds + (n0 + 2 * j + 0) * LDP + kr) =
            make_uint2(perm_lo(w1[j], w0[j]), perm_lo(w3[j], w2[j]));
        *(uint2*)(lds + (n0 + 2 * j + 1) * LDP + kr) =
            make_uint2(perm_hi(w1[j], w0[j]), perm_hi(w3[j], w2[j]));
    }
}

// ---- fragment loads (ISA 7.12.2 layouts) ----
// A 16x32 bf16: lane m=L&15, h=L>>4; elems e<8 -> K=8h+e ; e>=8 -> K=16+8h+(e-8)
__device__ __forceinline__ v16bf load_fragA(const unsigned short* rowblk, int lane) {
    const unsigned short* p = rowblk + (lane & 15) * LDP;
    const int h = (lane >> 4) << 3;                 // 0 or 8
    v8bf lo = *(const v8bf*)(p + h);
    v8bf hi = *(const v8bf*)(p + 16 + h);
    return __builtin_shufflevector(lo, hi, 0,1,2,3,4,5,6,7,8,9,10,11,12,13,14,15);
}
// B 32x16 bf16 (LDS stored [n][k]): lane n=L&15, h=L>>4; elems e<8 -> K=8h+e ; e>=8 -> K=16+8h+(e-8)
__device__ __forceinline__ v16bf load_fragB(const unsigned short* rowblk, int lane) {
    const unsigned short* p = rowblk + (lane & 15) * LDP + ((lane >> 4) << 3);
    v8bf lo = *(const v8bf*)(p);
    v8bf hi = *(const v8bf*)(p + 16);
    return __builtin_shufflevector(lo, hi, 0,1,2,3,4,5,6,7,8,9,10,11,12,13,14,15);
}

// 64x64 wave tile: 4 A frags reused across 4 B frags -> 16 wmma per K-step
__device__ __forceinline__ void mma_step(const unsigned short* ldsA, const unsigned short* ldsB,
                                         int wm, int wn, int lane, v8f acc[4][4]) {
    v16bf a[4];
#pragma unroll
    for (int mi = 0; mi < 4; ++mi)
        a[mi] = load_fragA(ldsA + (wm + mi * 16) * LDP, lane);
#pragma unroll
    for (int ni = 0; ni < 4; ++ni) {
        v16bf b = load_fragB(ldsB + (wn + ni * 16) * LDP, lane);
#pragma unroll
        for (int mi = 0; mi < 4; ++mi)
            acc[mi][ni] = __builtin_amdgcn_wmma_f32_16x16x32_bf16(
                false, a[mi], false, b, (short)0, acc[mi][ni], false, false);
    }
}

#define GEMM_PROLOGUE()                                                   \
    __shared__ __align__(16) unsigned short ldsA[128 * LDP];              \
    __shared__ __align__(16) unsigned short ldsB[256 * LDP];              \
    const int tid  = threadIdx.x;                                         \
    const int lane = tid & 31;                                            \
    const int wave = tid >> 5;                                            \
    const int wm   = (wave & 1) << 6;                                     \
    const int wn   = (wave >> 1) << 6;                                    \
    const int mtile = blockIdx.y << 7;                                    \
    const int ntile = blockIdx.x << 8;                                    \
    v8f acc[4][4];                                                        \
    {                                                                     \
        v8f z = { 0.f, 0.f, 0.f, 0.f, 0.f, 0.f, 0.f, 0.f };              \
        _Pragma("unroll")                                                 \
        for (int i = 0; i < 4; ++i)                                       \
            _Pragma("unroll")                                             \
            for (int j = 0; j < 4; ++j) acc[i][j] = z;                    \
    }

// Variadic so statement bodies containing commas survive preprocessing.
#define EPILOGUE_LOOP(...)                                                \
    _Pragma("unroll")                                                     \
    for (int mi = 0; mi < 4; ++mi)                                        \
        _Pragma("unroll")                                                 \
        for (int ni = 0; ni < 4; ++ni) {                                  \
            v8f v = acc[mi][ni];                                          \
            _Pragma("unroll")                                             \
            for (int r = 0; r < 8; ++r) {                                 \
                int gm = mtile + wm + (mi << 4) + ((lane >> 4) << 3) + r; \
                int gn = ntile + wn + (ni << 4) + (lane & 15);            \
                __VA_ARGS__                                               \
            }                                                             \
        }

// ===========================================================================
// Stage 1: q / kv 1x1-conv projections.
//   out(o,p) = sum_c W[o][c] * X[n][c][p] + bias[o]
//   q rows (s<3)  -> out_t[(n*3+s)*512 + t][f]      (t-major, f = cm*25+v)
//   v rows (s>=3) -> out_v[(n*3+s-3)*1600 + f][t]   (f-major; transposed for GEMM-3)
// ===========================================================================
__global__ void __launch_bounds__(256) k_proj(const float* __restrict__ W,
                                              const float* __restrict__ bias,
                                              const float* __restrict__ X, int M,
                                              unsigned short* __restrict__ out_t,
                                              unsigned short* __restrict__ out_v) {
    GEMM_PROLOGUE();
    const int nb = blockIdx.z;
    const float* Xb = X + (long)nb * 192 * 12800;
#pragma unroll 1
    for (int k0 = 0; k0 < 192; k0 += 32) {
        __syncthreads();
        stage_mk_f32(ldsA, W + k0, 192, mtile, M, tid);
        stage_kn_f32(ldsB, Xb + (long)k0 * 12800 + ntile, 12800, tid);
        if (k0 + 32 < 192) {   // prefetch next B tile (dominant stream)
            __builtin_prefetch(Xb + (long)(k0 + 32 + ((tid >> 5) << 2)) * 12800 +
                               ntile + ((tid & 31) << 3), 0, 1);
        }
        __syncthreads();
        mma_step(ldsA, ldsB, wm, wn, lane, acc);
    }
    EPILOGUE_LOOP({
        if (gm < M) {
            float val = v[r] + bias[gm];
            int sIdx = gm >> 6;
            int cm = gm & 63;
            int t = gn / 25;
            int vv = gn - t * 25;
            int f = cm * 25 + vv;
            unsigned short bv = f2bf(val);
            if (sIdx < 3)
                out_t[((long)(nb * 3 + sIdx) * 512 + t) * 1600 + f] = bv;
            else
                out_v[((long)(nb * 3 + sIdx - 3) * 1600 + f) * 512 + t] = bv;
        }
    })
}

// ===========================================================================
// Stage 2: attn = tanh(Q K^T / 1600) per (n,s). M=N=512, K=1600.
// ===========================================================================
__global__ void __launch_bounds__(256) k_score(const unsigned short* __restrict__ q,
                                               const unsigned short* __restrict__ k,
                                               unsigned short* __restrict__ attn) {
    GEMM_PROLOGUE();
    const long base = (long)blockIdx.z * 512 * 1600;
#pragma unroll 1
    for (int k0 = 0; k0 < 1600; k0 += 32) {
        __syncthreads();
        stage_mk_bf16<128>(ldsA, q + base + k0, 1600, mtile, 512, tid);
        stage_mk_bf16<256>(ldsB, k + base + k0, 1600, ntile, 512, tid);
        if (k0 + 32 < 1600) {
            __builtin_prefetch(q + base + (long)(mtile + (tid >> 1)) * 1600 + k0 + 32, 0, 1);
            __builtin_prefetch(k + base + (long)(ntile + (tid >> 1)) * 1600 + k0 + 32, 0, 1);
        }
        __syncthreads();
        mma_step(ldsA, ldsB, wm, wn, lane, acc);
    }
    const long obase = (long)blockIdx.z * 512 * 512;
    EPILOGUE_LOOP({
        float val = tanhf(v[r] * (1.0f / 1600.0f));
        attn[obase + (long)gm * 512 + gn] = f2bf(val);
    })
}

// ===========================================================================
// Stage 3: Y = attn * V per (n,s). M=512 (t), N=1600 (f), K=512 (q').
// Output scattered back to standard [n][C][T][V] bf16 layout.
// ===========================================================================
__global__ void __launch_bounds__(256) k_apply(const unsigned short* __restrict__ attn,
                                               const unsigned short* __restrict__ vatt,
                                               unsigned short* __restrict__ y) {
    GEMM_PROLOGUE();
    const int z = blockIdx.z;
    const long abase = (long)z * 512 * 512;
    const long vbase = (long)z * 1600 * 512;
#pragma unroll 1
    for (int k0 = 0; k0 < 512; k0 += 32) {
        __syncthreads();
        stage_mk_bf16<128>(ldsA, attn + abase + k0, 512, mtile, 512, tid);
        stage_mk_bf16<256>(ldsB, vatt + vbase + k0, 512, ntile, 1600, tid);
        if (k0 + 32 < 512) {
            __builtin_prefetch(attn + abase + (long)(mtile + (tid >> 1)) * 512 + k0 + 32, 0, 1);
            __builtin_prefetch(vatt + vbase + (long)(ntile + (tid >> 1)) * 512 + k0 + 32, 0, 1);
        }
        __syncthreads();
        mma_step(ldsA, ldsB, wm, wn, lane, acc);
    }
    const int nb = z / 3;
    const int s  = z - nb * 3;
    EPILOGUE_LOOP({
        if (gn < 1600) {
            int c  = (s << 6) + gn / 25;
            int vv = gn % 25;
            y[((long)(nb * 192 + c) * 512 + gm) * 25 + vv] = f2bf(v[r]);
        }
    })
}

// ===========================================================================
// Stage 4: out = LeakyReLU(x1 + BN(Wf*y + bf)). M=192, N=12800, K=192.
// ===========================================================================
__global__ void __launch_bounds__(256) k_out(const float* __restrict__ Wf,
                                             const float* __restrict__ bfv,
                                             const unsigned short* __restrict__ y,
                                             const float* __restrict__ x1,
                                             const float* __restrict__ gamma,
                                             const float* __restrict__ beta,
                                             const float* __restrict__ mean,
                                             const float* __restrict__ var,
                                             float* __restrict__ out) {
    GEMM_PROLOGUE();
    const int nb = blockIdx.z;
    const unsigned short* Yb = y + (long)nb * 192 * 12800;
#pragma unroll 1
    for (int k0 = 0; k0 < 192; k0 += 32) {
        __syncthreads();
        stage_mk_f32(ldsA, Wf + k0, 192, mtile, 192, tid);
        stage_kn_bf16(ldsB, Yb + (long)k0 * 12800 + ntile, 12800, tid);
        if (k0 + 32 < 192) {
            __builtin_prefetch(Yb + (long)(k0 + 32 + ((tid >> 5) << 2)) * 12800 +
                               ntile + ((tid & 31) << 3), 0, 1);
        }
        __syncthreads();
        mma_step(ldsA, ldsB, wm, wn, lane, acc);
    }
    EPILOGUE_LOOP({
        if (gm < 192) {
            float inv = gamma[gm] * rsqrtf(var[gm] + 1e-5f);
            float val = (v[r] + bfv[gm]) * inv + (beta[gm] - mean[gm] * inv);
            long idx = ((long)nb * 192 + gm) * 12800 + gn;
            val += x1[idx];
            out[idx] = val >= 0.f ? val : 0.1f * val;
        }
    })
}

// ===========================================================================
// Host launcher. Workspace layout (bytes), needs ~261 MB:
//   [0)            q_att  bf16 [16*3][512][1600]   (aliased by y after stage 2)
//   [78,643,200)   k_att  bf16 [16*3][512][1600]
//   [157,286,400)  v_att  bf16 [16*3][1600][512]
//   [235,929,600)  attn   bf16 [48][512][512]
// ===========================================================================
extern "C" void kernel_launch(void* const* d_in, const int* in_sizes, int n_in,
                              void* d_out, int out_size, void* d_ws, size_t ws_size,
                              hipStream_t stream) {
    const float* x1   = (const float*)d_in[0];
    const float* x2   = (const float*)d_in[1];
    const float* Wq   = (const float*)d_in[2];
    const float* bq   = (const float*)d_in[3];
    const float* Wkv  = (const float*)d_in[4];
    const float* bkv  = (const float*)d_in[5];
    const float* Wf   = (const float*)d_in[6];
    const float* bf_  = (const float*)d_in[7];
    const float* gam  = (const float*)d_in[8];
    const float* bet  = (const float*)d_in[9];
    const float* mean = (const float*)d_in[10];
    const float* var  = (const float*)d_in[11];

    char* base = (char*)d_ws;
    unsigned short* q_att = (unsigned short*)(base);
    unsigned short* k_att = (unsigned short*)(base + 78643200L);
    unsigned short* v_att = (unsigned short*)(base + 157286400L);
    unsigned short* attn  = (unsigned short*)(base + 235929600L);
    unsigned short* y_std = q_att;   // q dead after stage 2; same size

    dim3 blk(256);
    // Stage 1: projections (q from x1; k,v from x2)
    k_proj<<<dim3(50, 2, 16), blk, 0, stream>>>(Wq,  bq,  x1, 192, q_att, q_att);
    k_proj<<<dim3(50, 3, 16), blk, 0, stream>>>(Wkv, bkv, x2, 384, k_att, v_att);
    // Stage 2: attention scores, tanh-normalized
    k_score<<<dim3(2, 4, 48), blk, 0, stream>>>(q_att, k_att, attn);
    // Stage 3: apply attention to V
    k_apply<<<dim3(7, 4, 48), blk, 0, stream>>>(attn, v_att, y_std);
    // Stage 4: output projection + BN + residual + LeakyReLU
    k_out<<<dim3(50, 2, 16), blk, 0, stream>>>(Wf, bf_, y_std, x1, gam, bet, mean,
                                               var, (float*)d_out);
}
